// Bottleneck_DynamicConv_55070070669647
// MI455X (gfx1250) — compile-verified, matrix operand order checked by reading, not run on hardware
//
#include <hip/hip_runtime.h>

// ---------------------------------------------------------------------------
// Bottleneck + DynamicConv (CondConv) for MI455X (gfx1250, wave32, WMMA+TDM)
//
// B=16, C=256, H=W=64, E=4, two 3x3 pad-1 convs as implicit GEMMs:
//   M=256 (Cout), N=4096 (HW), K=2304 ordered TAP-MAJOR: k = tap*256 + ci
// f16 operands (pre-converted), f32 accumulate via v_wmma_f32_16x16x32_f16.
// A tiles staged LDS via Tensor Data Mover (tensor_load_to_lds + TENSORcnt).
// ---------------------------------------------------------------------------

typedef __attribute__((ext_vector_type(16))) _Float16 v16h;
typedef __attribute__((ext_vector_type(8)))  _Float16 v8h;
typedef __attribute__((ext_vector_type(8)))  float    v8f;
typedef __attribute__((ext_vector_type(4)))  unsigned int u32x4;
typedef __attribute__((ext_vector_type(8)))  int          i32x8;
typedef __attribute__((ext_vector_type(4)))  int          i32x4;

#define BATCH   16
#define CCH     256
#define HW      4096      // 64*64
#define KDIM    2304      // 9*256, tap-major: k = tap*256 + ci
#define NEXP    4
#define WELEM   589824    // C*C*9 per-sample kernel elements
#define EPSBN   1e-5f

#define BLK_M   128
#define BLK_N   128
#define BLK_K   32
#define LDSW    40        // padded LDS row stride in halves (64B data + 16B pad)
#define NTHR    256       // 8 waves

#if defined(__AMDGCN__) && __has_builtin(__builtin_amdgcn_tensor_load_to_lds)
#define USE_TDM 1
#else
#define USE_TDM 0
#endif

__device__ __forceinline__ float silu_f(float v) {
    return v * (1.0f / (1.0f + __expf(-v)));
}

#if USE_TDM
// Issue one TDM 2D tile load: 128 rows x 32 f16, row stride KDIM elements,
// into LDS at lds_addr with 16B padding after each 64B row (LDS stride 80B).
__device__ __forceinline__ void tdm_load_tile(unsigned lds_addr, const _Float16* gsrc) {
    unsigned long long ga = (unsigned long long)(size_t)gsrc;
    u32x4 g0;
    g0[0] = 1u;                                            // count=1, user, no gather
    g0[1] = lds_addr;                                      // bits 63:32  lds_addr
    g0[2] = (unsigned)(ga & 0xffffffffu);                  // bits 95:64  global_addr lo
    g0[3] = (unsigned)((ga >> 32) & 0x01ffffffu)           // bits 120:96 global_addr hi
          | (2u << 30);                                    // type = 2 ("image")
    i32x8 g1;
    g1[0] = (1 << 16)                                      // data_size = 1 -> 2 bytes
          | (1 << 20)                                      // pad_enable
          | (3 << 22)                                      // pad_interval: 3 -> every 16 DW (64B)
          | (3 << 25);                                     // pad_amount:   3 -> 4 DW (16B)
    g1[1] = (int)((KDIM & 0xffffu) << 16);                 // tensor_dim0[15:0] @ bits 63:48
    g1[2] = (int)((KDIM >> 16) | ((CCH & 0xffffu) << 16)); // dim0 hi | tensor_dim1 lo
    g1[3] = (int)((CCH >> 16) | (BLK_K << 16));            // dim1 hi | tile_dim0 = 32
    g1[4] = (int)(BLK_M);                                  // tile_dim1 = 128, tile_dim2 = 0
    g1[5] = (int)KDIM;                                     // tensor_dim0_stride lo32
    g1[6] = 0;                                             // stride hi | dim1_stride lo
    g1[7] = 0;
    i32x4 z4 = {0, 0, 0, 0};
    i32x8 z8 = {0, 0, 0, 0, 0, 0, 0, 0};
    __builtin_amdgcn_tensor_load_to_lds(g0, g1, z4, z4, z8, 0);
}
#endif

// ---------------------------------------------------------------------------
// MODE 0: cv1 : A = w1T (f16, [co][tap*256+ci]), In = xh, Out -> yh (f16), BN1+SiLU
// MODE 1: cv2 : A = kern (f16, per-b),           In = yh, Out -> d_out (f32),
//               BN2+SiLU+residual(x)
// ---------------------------------------------------------------------------
template <int MODE>
__global__ __launch_bounds__(NTHR)
void conv_gemm_wmma(const _Float16* __restrict__ Abase,
                    const _Float16* __restrict__ In,
                    float* __restrict__ OutF,
                    _Float16* __restrict__ OutH,
                    const float* __restrict__ xres,
                    const float* __restrict__ bn_g,
                    const float* __restrict__ bn_b,
                    const float* __restrict__ bn_m,
                    const float* __restrict__ bn_v) {
    __shared__ _Float16 Atile[BLK_M][LDSW];   // 10 KB
    __shared__ _Float16 Btile[BLK_N][LDSW];   // 10 KB  ([pixel][k-in-block])

    const int tid  = threadIdx.x;
    const int lane = tid & 31;
    const int wave = tid >> 5;
    const int wave_m = wave & 1;   // 2 waves along M -> 64 rows each
    const int wave_n = wave >> 1;  // 4 waves along N -> 32 cols each

    const int bid   = blockIdx.x;            // b*64 + tm*32 + tn
    const int b     = bid >> 6;
    const int tm    = (bid >> 5) & 1;
    const int tn    = bid & 31;
    const int tileM = tm * BLK_M;
    const int tileN = tn * BLK_N;

    v8f acc[4][2];
    #pragma unroll
    for (int i = 0; i < 4; ++i)
        #pragma unroll
        for (int j = 0; j < 2; ++j)
            acc[i][j] = (v8f)0.0f;

    // --- B staging coords: 32 k-columns x 8 pixel-groups of 16 ---
    const int kk   = tid >> 3;          // 0..31 : k column within block (= ci offset)
    const int pg   = tid & 7;           // 0..7  : pixel group
    const int ph   = pg >> 2;           // 0..1  : local image row
    const int pw0  = (pg & 3) * 16;     // 0,16,32,48
    const int trow = (tileN >> 6) + ph; // global image row of this thread's pixels
    const _Float16* Inb = In + ((size_t)b * CCH) * (size_t)HW;

#if !USE_TDM
    const int srow  = tid >> 1;
    const int scol0 = (tid & 1) * 16;
#endif
    const _Float16* Ab = Abase + (MODE == 1 ? (size_t)b * WELEM : (size_t)0)
                               + (size_t)tileM * KDIM;

    // fragment K offset (documented 16-bit A layout): lanes 0-15 read K[0..7]
    // and K[16..23], lanes 16-31 read K[8..15] and K[24..31]
    const int lkoff = (lane >> 4) * 8;
    const int mrow  = lane & 15;

#if USE_TDM
    const unsigned lds_a = (unsigned)(size_t)(&Atile[0][0]);
#endif

    for (int k0 = 0; k0 < KDIM; k0 += BLK_K) {
        // ---- stage A tile (TDM async DMA, wave 0 issues) ----
#if USE_TDM
        if (wave == 0) {
            tdm_load_tile(lds_a, Ab + k0);
        }
#else
        {
            const v8h* src = (const v8h*)(Ab + (size_t)srow * KDIM + k0 + scol0);
            *(v8h*)&Atile[srow][scol0]     = src[0];
            *(v8h*)&Atile[srow][scol0 + 8] = src[1];
        }
#endif

        // ---- stage B tile: tap-major im2col, no div/mod in the loop ----
        {
            const int tap = k0 >> 8;            // 0..8, constant within block
            const int dh  = tap / 3 - 1;
            const int dw  = tap % 3 - 1;
            const int ci  = (k0 & 255) + kk;
            const int hh  = trow + dh;
            const bool vh = (unsigned)hh < 64u;
            const _Float16* row = Inb + ((size_t)ci << 12) + (hh << 6);
            if (vh) __builtin_prefetch(row + (BLK_K << 12), 0, 0);  // next ci block
            #pragma unroll
            for (int j = 0; j < 16; ++j) {
                const int ww = pw0 + j + dw;
                _Float16 v = (_Float16)0.0f;
                if (vh && (unsigned)ww < 64u) v = row[ww];
                Btile[ph * 64 + pw0 + j][kk] = v;
            }
        }

#if USE_TDM
        __builtin_amdgcn_s_wait_tensorcnt((short)0);
#endif
        __syncthreads();

        // ---- fragment loads + 8 WMMAs ----
        v16h afrag[4], bfrag[2];
        #pragma unroll
        for (int i = 0; i < 4; ++i) {
            const int m = wave_m * 64 + i * 16 + mrow;
            union { v16h v; v8h h[2]; } u;
            u.h[0] = *(const v8h*)&Atile[m][lkoff];
            u.h[1] = *(const v8h*)&Atile[m][lkoff + 16];
            afrag[i] = u.v;
        }
        #pragma unroll
        for (int j = 0; j < 2; ++j) {
            const int n = wave_n * 32 + j * 16 + mrow;
            union { v16h v; v8h h[2]; } u;
            u.h[0] = *(const v8h*)&Btile[n][lkoff];
            u.h[1] = *(const v8h*)&Btile[n][lkoff + 16];
            bfrag[j] = u.v;
        }
        #pragma unroll
        for (int i = 0; i < 4; ++i)
            #pragma unroll
            for (int j = 0; j < 2; ++j)
                acc[i][j] = __builtin_amdgcn_wmma_f32_16x16x32_f16(
                    false, afrag[i], false, bfrag[j], (short)0, acc[i][j], false, false);

        __syncthreads();
    }

    // ---- epilogue: BN + SiLU; MODE0 -> f16 yh, MODE1 -> f32 out + residual ----
    const int nin   = lane & 15;
    const int mbase = (lane >> 4) * 8;   // C/D layout: vgpr r -> M=r (lo), r+8 (hi)
    #pragma unroll
    for (int i = 0; i < 4; ++i) {
        #pragma unroll
        for (int j = 0; j < 2; ++j) {
            const int nglob = tileN + wave_n * 32 + j * 16 + nin;
            #pragma unroll
            for (int r = 0; r < 8; ++r) {
                const int co = tileM + wave_m * 64 + i * 16 + mbase + r;
                const float s = bn_g[co] * rsqrtf(bn_v[co] + EPSBN);
                const float val = (acc[i][j][r] - bn_m[co]) * s + bn_b[co];
                const float o = silu_f(val);
                const size_t oidx = ((size_t)b * CCH + co) * (size_t)HW + nglob;
                if (MODE == 0) OutH[oidx] = (_Float16)o;
                else           OutF[oidx] = o + xres[oidx];
            }
        }
    }
}

// ---------------------------------------------------------------------------
// x (f32) -> xh (f16), 4 elements per thread
// ---------------------------------------------------------------------------
__global__ __launch_bounds__(256)
void cvt_x_kernel(const float* __restrict__ x, _Float16* __restrict__ xh) {
    const size_t base = ((size_t)blockIdx.x * 256 + threadIdx.x) * 4;
    #pragma unroll
    for (int j = 0; j < 4; ++j) xh[base + j] = (_Float16)x[base + j];
}

// ---------------------------------------------------------------------------
// w1 [co][ci][tap] f32 -> w1T [co][tap*256+ci] f16 (tap-major K order)
// ---------------------------------------------------------------------------
__global__ __launch_bounds__(256)
void cvt_w1_kernel(const float* __restrict__ w1, _Float16* __restrict__ w1T) {
    const int id = blockIdx.x * 256 + threadIdx.x;   // 0 .. 589823 (src index)
    const int co  = id / KDIM;
    const int r   = id - co * KDIM;
    const int ci  = r / 9;
    const int tap = r - ci * 9;
    w1T[(size_t)co * KDIM + tap * CCH + ci] = (_Float16)w1[id];
}

// ---------------------------------------------------------------------------
// Global average pool over f16 y: one block per (b,c) row of 4096.
// ---------------------------------------------------------------------------
__global__ __launch_bounds__(256)
void pool_kernel(const _Float16* __restrict__ yh, float* __restrict__ pooled) {
    __shared__ float red[256];
    const size_t bc = blockIdx.x;
    const _Float16* row = yh + bc * (size_t)HW;
    float s = 0.0f;
    for (int i = threadIdx.x; i < HW; i += 256) s += (float)row[i];
    red[threadIdx.x] = s;
    __syncthreads();
    for (int st = 128; st > 0; st >>= 1) {
        if (threadIdx.x < st) red[threadIdx.x] += red[threadIdx.x + st];
        __syncthreads();
    }
    if (threadIdx.x == 0) pooled[bc] = red[0] * (1.0f / (float)HW);
}

// ---------------------------------------------------------------------------
// Routing gates: sigmoid(pooled @ wr^T + br), 16x4, one block.
// ---------------------------------------------------------------------------
__global__ __launch_bounds__(64)
void routing_kernel(const float* __restrict__ pooled,
                    const float* __restrict__ wr,
                    const float* __restrict__ br,
                    float* __restrict__ routing) {
    const int t = threadIdx.x;
    if (t >= BATCH * NEXP) return;
    const int b = t >> 2;
    const int e = t & 3;
    float s = br[e];
    for (int c = 0; c < CCH; ++c)
        s += pooled[b * CCH + c] * wr[e * CCH + c];
    routing[t] = 1.0f / (1.0f + __expf(-s));
}

// ---------------------------------------------------------------------------
// Expert mix -> per-sample kernel bank, f16, tap-major layout:
//   kern[b][co][tap*256+ci] = sum_e routing[b,e] * w_e[e][co*2304 + ci*9 + tap]
// ---------------------------------------------------------------------------
__global__ __launch_bounds__(256)
void mixkern_kernel(const float* __restrict__ routing,
                    const float* __restrict__ w_e,
                    _Float16* __restrict__ kern) {
    const size_t id = (size_t)blockIdx.x * 256 + threadIdx.x;  // dst index
    const int    b   = (int)(id / WELEM);
    const int    rem = (int)(id - (size_t)b * WELEM);
    const int    co  = rem / KDIM;
    const int    r2  = rem - co * KDIM;
    const int    tap = r2 >> 8;
    const int    ci  = r2 & 255;
    const size_t src = (size_t)co * KDIM + ci * 9 + tap;
    float s = routing[b * 4 + 0] * w_e[0 * (size_t)WELEM + src]
            + routing[b * 4 + 1] * w_e[1 * (size_t)WELEM + src]
            + routing[b * 4 + 2] * w_e[2 * (size_t)WELEM + src]
            + routing[b * 4 + 3] * w_e[3 * (size_t)WELEM + src];
    kern[id] = (_Float16)s;
}

// ---------------------------------------------------------------------------
// Host-side launch
// ---------------------------------------------------------------------------
extern "C" void kernel_launch(void* const* d_in, const int* in_sizes, int n_in,
                              void* d_out, int out_size, void* d_ws, size_t ws_size,
                              hipStream_t stream) {
    const float* x     = (const float*)d_in[0];
    const float* w1    = (const float*)d_in[1];
    const float* bn1_g = (const float*)d_in[2];
    const float* bn1_b = (const float*)d_in[3];
    const float* bn1_m = (const float*)d_in[4];
    const float* bn1_v = (const float*)d_in[5];
    const float* wr    = (const float*)d_in[6];
    const float* br    = (const float*)d_in[7];
    const float* w_e   = (const float*)d_in[8];
    const float* bn2_g = (const float*)d_in[9];
    const float* bn2_b = (const float*)d_in[10];
    const float* bn2_m = (const float*)d_in[11];
    const float* bn2_v = (const float*)d_in[12];
    float* out = (float*)d_out;

    // workspace layout (f16-centric)
    const size_t NX = (size_t)BATCH * CCH * HW;   // 16,777,216
    _Float16* xh      = (_Float16*)d_ws;          // 32 MB
    _Float16* yh      = xh + NX;                  // 32 MB
    _Float16* w1T     = yh + NX;                  // 1.125 MB
    _Float16* kern    = w1T + (size_t)CCH * KDIM; // 18 MB
    float*    pooled  = (float*)(kern + (size_t)BATCH * WELEM);
    float*    routing = pooled + BATCH * CCH;

    const int gemm_grid = BATCH * (CCH / BLK_M) * (HW / BLK_N);  // 1024

    // precision conversion + weight permutation (tap-major K)
    cvt_x_kernel<<<(int)(NX / 4 / 256), 256, 0, stream>>>(x, xh);
    cvt_w1_kernel<<<(CCH * KDIM) / 256, 256, 0, stream>>>(w1, w1T);

    // cv1: conv + BN1 + SiLU -> yh (f16)
    conv_gemm_wmma<0><<<gemm_grid, NTHR, 0, stream>>>(
        w1T, xh, nullptr, yh, nullptr, bn1_g, bn1_b, bn1_m, bn1_v);

    // routing gates
    pool_kernel<<<BATCH * CCH, 256, 0, stream>>>(yh, pooled);
    routing_kernel<<<1, 64, 0, stream>>>(pooled, wr, br, routing);

    // per-sample kernel bank (f16, tap-major)
    mixkern_kernel<<<(int)(((size_t)BATCH * WELEM) / 256), 256, 0, stream>>>(
        routing, w_e, kern);

    // cv2: per-sample conv + BN2 + SiLU + residual -> out (f32)
    conv_gemm_wmma<1><<<gemm_grid, NTHR, 0, stream>>>(
        kern, yh, out, nullptr, x, bn2_g, bn2_b, bn2_m, bn2_v);
}
